// PeakTransformerImpClassifier_62191126446774
// MI455X (gfx1250) — matrix operand order repaired and checked by reading, not compiled, not run on hardware
//
#include <hip/hip_runtime.h>
#include <hip/hip_bf16.h>
#include <math.h>

// ---------------------------------------------------------------------------
// Problem constants (match the reference)
// ---------------------------------------------------------------------------
#define B_   4
#define N_   1024
#define PD_  8
#define SD_  16
#define D_   160
#define H_   8
#define HD_  20
#define L_   5
#define FF_  640
#define WIN_ 32
#define S_   (N_ + 1)          // 1025 tokens
#define M_   (B_ * S_)         // 4100 rows in the residual stream
#define MP_  4112              // M_ padded to a multiple of 16 (257 tiles)
#define MN_  (B_ * N_)         // 4096 head rows (already a multiple of 16)

typedef __attribute__((ext_vector_type(16))) _Float16 v16h;
typedef __attribute__((ext_vector_type(8)))  _Float16 v8h;
typedef __attribute__((ext_vector_type(8)))  float    v8f;

__device__ __forceinline__ v16h cat16(v8h lo, v8h hi)
{
    return __builtin_shufflevector(lo, hi, 0,1,2,3,4,5,6,7,8,9,10,11,12,13,14,15);
}

// ---------------------------------------------------------------------------
// Weight transpose + f32->f16 convert:  W[K,Nn] (row major) -> Wt[Nn,K]
// blockIdx.z selects the layer slice.
// ---------------------------------------------------------------------------
__global__ void wcvt_kernel(const float* __restrict__ W, _Float16* __restrict__ Wt,
                            int K, int Nn)
{
    const float* src = W  + (size_t)blockIdx.z * K * Nn;
    _Float16*    dst = Wt + (size_t)blockIdx.z * K * Nn;
    int tid = blockIdx.x * blockDim.x + threadIdx.x;
    if (tid >= K * Nn) return;
    int n = tid / K;
    int k = tid - n * K;
    dst[(size_t)n * K + k] = (_Float16)src[(size_t)k * Nn + n];
}

// ---------------------------------------------------------------------------
// Token embedding: peak projection + token types + rank emb + FiLM from
// spectrum, and the global spectrum token at s==0.  Output: raw tokens f32.
// ---------------------------------------------------------------------------
__global__ void embed_kernel(const float* __restrict__ peak, const float* __restrict__ spec,
                             const float* __restrict__ Wp,  const float* __restrict__ bp,
                             const float* __restrict__ Ws,  const float* __restrict__ bs,
                             const float* __restrict__ peak_tt, const float* __restrict__ spec_tt,
                             const float* __restrict__ rank_emb,
                             const float* __restrict__ Wsc, const float* __restrict__ bsc,
                             const float* __restrict__ Wsh, const float* __restrict__ bsh,
                             float* __restrict__ tok)
{
    int tid = blockIdx.x * blockDim.x + threadIdx.x;
    if (tid >= M_ * D_) return;
    int r = tid / D_, d = tid - r * D_;
    int b = r / S_,  s = r - b * S_;
    const float* sp = spec + b * SD_;
    float val;
    if (s == 0) {
        val = bs[d] + spec_tt[d];
        #pragma unroll
        for (int k = 0; k < SD_; ++k) val += sp[k] * Ws[k * D_ + d];
    } else {
        int n = s - 1;
        const float* pf = peak + (size_t)(b * N_ + n) * PD_;
        val = bp[d] + peak_tt[d] + rank_emb[(size_t)n * D_ + d];
        #pragma unroll
        for (int k = 0; k < PD_; ++k) val += pf[k] * Wp[k * D_ + d];
        float sc = bsc[d], sh = bsh[d];
        #pragma unroll
        for (int k = 0; k < SD_; ++k) { sc += sp[k] * Wsc[k * D_ + d]; sh += sp[k] * Wsh[k * D_ + d]; }
        sc = tanhf(sc);
        val = val * (1.f + 0.1f * sc) + sh;
    }
    tok[(size_t)r * D_ + d] = val;
}

// ---------------------------------------------------------------------------
// LayerNorm over D=160: one wave32 per row, 5 elements per lane,
// shfl_xor tree reductions.  OutT = float (in-place residual) or _Float16
// (GEMM A-operand).
// ---------------------------------------------------------------------------
template <typename OutT>
__global__ void ln_kernel(const float* __restrict__ src, OutT* __restrict__ dst,
                          const float* __restrict__ g, const float* __restrict__ bb,
                          int rows)
{
    int row = blockIdx.x;
    if (row >= rows) return;
    int l = threadIdx.x;
    const float* p = src + (size_t)row * D_;
    float v[5];
    float s = 0.f;
    #pragma unroll
    for (int j = 0; j < 5; ++j) { v[j] = p[l + 32 * j]; s += v[j]; }
    #pragma unroll
    for (int m = 16; m > 0; m >>= 1) s += __shfl_xor(s, m, 32);
    float mean = s * (1.f / (float)D_);
    float sq = 0.f;
    #pragma unroll
    for (int j = 0; j < 5; ++j) { float t = v[j] - mean; sq += t * t; }
    #pragma unroll
    for (int m = 16; m > 0; m >>= 1) sq += __shfl_xor(sq, m, 32);
    float rstd = rsqrtf(sq * (1.f / (float)D_) + 1e-5f);
    OutT* q = dst + (size_t)row * D_;
    #pragma unroll
    for (int j = 0; j < 5; ++j) {
        int d = l + 32 * j;
        q[d] = (OutT)((v[j] - mean) * rstd * g[d] + bb[d]);
    }
}

// ---------------------------------------------------------------------------
// WMMA GEMM:  C[M,Nn] = A[M,K](f16,row-major) * Wt[Nn,K](f16)^T + bias
// One wave per 16x(16*NB) tile; K in steps of 32 via v_wmma_f32_16x16x32_f16.
// The A k-step load is reused for NB WMMAs (NB=2 -> 3 b128 loads per WMMA).
//   A lane layout (ISA 7.12.2, 16-bit A 16x32): lanes 0-15 K0-7 & K16-23,
//   lanes 16-31 K8-15 & K24-31 -> two contiguous v8h loads at +0 and +16.
//   B lane layout (same k order as A): lanes 0-15 K0-7 & K16-23,
//   lanes 16-31 K8-15 & K24-31 -> two v8h loads at +0 and +16 from the pre-transposed weight.
// MODE 0: f32 out (+bias).  MODE 1: f32 residual accumulate (+bias).
// MODE 2: exact GELU -> f16 out.
// ---------------------------------------------------------------------------
template <int MODE, int NB>
__global__ __launch_bounds__(32)
void gemm_kernel(const _Float16* __restrict__ A, int lda,
                 const _Float16* __restrict__ Bw,
                 const float* __restrict__ bias,
                 int Mrows, int K,
                 float* __restrict__ outF, _Float16* __restrict__ outH, int ldo)
{
    int l      = threadIdx.x;
    int m0     = blockIdx.x * 16;
    int n0     = blockIdx.y * (16 * NB);
    int lane15 = l & 15;
    int hi     = l >> 4;                       // 0 for lanes 0-15, 1 for 16-31
    const _Float16* arow = A + (size_t)(m0 + lane15) * lda + hi * 8;
    const _Float16* bcol[NB];
    #pragma unroll
    for (int j = 0; j < NB; ++j)
        bcol[j] = Bw + (size_t)(n0 + j * 16 + lane15) * K + hi * 8;
    v8f c[NB] = {};
    for (int k0 = 0; k0 < K; k0 += 32) {
        v16h a = cat16(*(const v8h*)(arow + k0), *(const v8h*)(arow + k0 + 16));
        #pragma unroll
        for (int j = 0; j < NB; ++j) {
            v16h bv = cat16(*(const v8h*)(bcol[j] + k0), *(const v8h*)(bcol[j] + k0 + 16));
            c[j] = __builtin_amdgcn_wmma_f32_16x16x32_f16(
                       false, a, false, bv, (short)0, c[j], false, false);
        }
    }
    #pragma unroll
    for (int j = 0; j < NB; ++j) {
        int n  = n0 + j * 16 + lane15;
        float bn = bias[n];
        #pragma unroll
        for (int r = 0; r < 8; ++r) {
            int m = m0 + r + hi * 8;           // C layout: lanes16-31 hold M=r+8
            if (m < Mrows) {
                float val = c[j][r] + bn;
                if (MODE == 0)      outF[(size_t)m * ldo + n]  = val;
                else if (MODE == 1) outF[(size_t)m * ldo + n] += val;
                else                outH[(size_t)m * ldo + n] =
                    (_Float16)(0.5f * val * (1.f + erff(val * 0.70710678f)));
            }
        }
    }
}

// ---------------------------------------------------------------------------
// Fused Q/K/V projection: one wave computes the same 16x16 tile for all three
// weight matrices, reusing each A k-step load 3x (and 1 launch instead of 3).
// K = lda = ldo = D_ (160).
// ---------------------------------------------------------------------------
__global__ __launch_bounds__(32)
void qkv_kernel(const _Float16* __restrict__ A,
                const _Float16* __restrict__ Wqt, const _Float16* __restrict__ Wkt,
                const _Float16* __restrict__ Wvt,
                const float* __restrict__ bq, const float* __restrict__ bk,
                const float* __restrict__ bv, int Mrows,
                float* __restrict__ Qo, float* __restrict__ Ko, float* __restrict__ Vo)
{
    int l      = threadIdx.x;
    int m0     = blockIdx.x * 16;
    int n0     = blockIdx.y * 16;
    int lane15 = l & 15;
    int hi     = l >> 4;
    const _Float16* arow = A + (size_t)(m0 + lane15) * D_ + hi * 8;
    size_t boff = (size_t)(n0 + lane15) * D_ + hi * 8;
    const _Float16* bq_p = Wqt + boff;
    const _Float16* bk_p = Wkt + boff;
    const _Float16* bv_p = Wvt + boff;
    v8f cq = {}, ck = {}, cv = {};
    for (int k0 = 0; k0 < D_; k0 += 32) {
        v16h a = cat16(*(const v8h*)(arow + k0), *(const v8h*)(arow + k0 + 16));
        v16h wq = cat16(*(const v8h*)(bq_p + k0), *(const v8h*)(bq_p + k0 + 16));
        cq = __builtin_amdgcn_wmma_f32_16x16x32_f16(false, a, false, wq, (short)0, cq, false, false);
        v16h wk = cat16(*(const v8h*)(bk_p + k0), *(const v8h*)(bk_p + k0 + 16));
        ck = __builtin_amdgcn_wmma_f32_16x16x32_f16(false, a, false, wk, (short)0, ck, false, false);
        v16h wv = cat16(*(const v8h*)(bv_p + k0), *(const v8h*)(bv_p + k0 + 16));
        cv = __builtin_amdgcn_wmma_f32_16x16x32_f16(false, a, false, wv, (short)0, cv, false, false);
    }
    int n = n0 + lane15;
    float bnq = bq[n], bnk = bk[n], bnv = bv[n];
    #pragma unroll
    for (int r = 0; r < 8; ++r) {
        int m = m0 + r + hi * 8;
        if (m < Mrows) {
            size_t o = (size_t)m * D_ + n;
            Qo[o] = cq[r] + bnq;
            Ko[o] = ck[r] + bnk;
            Vo[o] = cv[r] + bnv;
        }
    }
}

// ---------------------------------------------------------------------------
// Banded attention with global token + m/z log1p bias, streaming softmax.
// One thread per (b, head, query).  q==0 attends to everything; q>0 attends
// to key 0 plus tokens in [q-32, q+32].  Output written as f16 (next GEMM A).
// ---------------------------------------------------------------------------
__global__ void attn_kernel(const float* __restrict__ Q, const float* __restrict__ Kb,
                            const float* __restrict__ V, const float* __restrict__ peak,
                            const unsigned char* __restrict__ pad,
                            _Float16* __restrict__ O)
{
    int tid = blockIdx.x * blockDim.x + threadIdx.x;
    if (tid >= B_ * H_ * S_) return;
    int b   = tid / (H_ * S_);
    int rem = tid - b * H_ * S_;
    int h   = rem / S_;
    int qi  = rem - h * S_;

    size_t rbase = (size_t)(b * S_ + qi) * D_ + h * HD_;
    float qv[HD_];
    #pragma unroll
    for (int d = 0; d < HD_; ++d) qv[d] = Q[rbase + d];
    const float inv = 0.2236067977f;           // 1/sqrt(20)

    // key 0 (bias 0, never padded) seeds the online softmax
    size_t kb0 = (size_t)(b * S_) * D_ + h * HD_;
    float m;
    {
        float s = 0.f;
        #pragma unroll
        for (int d = 0; d < HD_; ++d) s += qv[d] * Kb[kb0 + d];
        m = s * inv;
    }
    float lsum = 1.f;
    float acc[HD_];
    #pragma unroll
    for (int d = 0; d < HD_; ++d) acc[d] = V[kb0 + d];

    int lo, hi2;
    if (qi == 0) { lo = 1; hi2 = N_; }
    else {
        lo  = (qi - WIN_ < 1)  ? 1  : qi - WIN_;
        hi2 = (qi + WIN_ > N_) ? N_ : qi + WIN_;
    }
    float mzq = (qi > 0) ? peak[((size_t)b * N_ + (qi - 1)) * PD_ + (PD_ - 1)] : 0.f;

    for (int ki = lo; ki <= hi2; ++ki) {
        if (pad[b * N_ + (ki - 1)]) continue;
        size_t kb2 = (size_t)(b * S_ + ki) * D_ + h * HD_;
        float s = 0.f;
        #pragma unroll
        for (int d = 0; d < HD_; ++d) s += qv[d] * Kb[kb2 + d];
        float bias = 0.f;
        if (qi > 0) {
            float mzk = peak[((size_t)b * N_ + (ki - 1)) * PD_ + (PD_ - 1)];
            bias = -0.25f * log1pf(fabsf(mzq - mzk));
        }
        s = s * inv + bias;
        float mn = fmaxf(m, s);
        float cf = __expf(m - mn);
        float ps = __expf(s - mn);
        lsum = lsum * cf + ps;
        #pragma unroll
        for (int d = 0; d < HD_; ++d) acc[d] = acc[d] * cf + ps * V[kb2 + d];
        m = mn;
    }
    float invl = 1.f / lsum;
    #pragma unroll
    for (int d = 0; d < HD_; ++d) O[rbase + d] = (_Float16)(acc[d] * invl);
}

// ---------------------------------------------------------------------------
// feat = concat(tokens[1:], broadcast(tokens[:1]))  ->  [B*N, 2D] f16
// ---------------------------------------------------------------------------
__global__ void feat_kernel(const _Float16* __restrict__ x, _Float16* __restrict__ feat)
{
    int tid = blockIdx.x * blockDim.x + threadIdx.x;
    if (tid >= MN_ * 2 * D_) return;
    int r = tid / (2 * D_);
    int c = tid - r * 2 * D_;
    int b = r / N_, n = r - b * N_;
    _Float16 v;
    if (c < D_) v = x[(size_t)(b * S_ + 1 + n) * D_ + c];
    else        v = x[(size_t)(b * S_) * D_ + (c - D_)];
    feat[(size_t)r * 2 * D_ + c] = v;
}

// ---------------------------------------------------------------------------
// Head second matmul (Nn=1 -> dot product) + padding mask
// ---------------------------------------------------------------------------
__global__ void head_kernel(const _Float16* __restrict__ hid, const float* __restrict__ Wh2,
                            const float* __restrict__ bh2, const unsigned char* __restrict__ pad,
                            float* __restrict__ out)
{
    int tid = blockIdx.x * blockDim.x + threadIdx.x;
    if (tid >= MN_) return;
    float s = bh2[0];
    const _Float16* p = hid + (size_t)tid * D_;
    #pragma unroll 8
    for (int d = 0; d < D_; ++d) s += (float)p[d] * Wh2[d];
    out[tid] = pad[tid] ? 0.f : s;
}

// ---------------------------------------------------------------------------
// Launch
// ---------------------------------------------------------------------------
extern "C" void kernel_launch(void* const* d_in, const int* in_sizes, int n_in,
                              void* d_out, int out_size, void* d_ws, size_t ws_size,
                              hipStream_t stream)
{
    const float* peak    = (const float*)d_in[0];
    const float* spec    = (const float*)d_in[1];
    const unsigned char* pad = (const unsigned char*)d_in[2];
    const float* Wp  = (const float*)d_in[3];  const float* bp  = (const float*)d_in[4];
    const float* Ws  = (const float*)d_in[5];  const float* bs  = (const float*)d_in[6];
    const float* ptt = (const float*)d_in[7];  const float* stt = (const float*)d_in[8];
    const float* remb= (const float*)d_in[9];
    const float* Wsc = (const float*)d_in[10]; const float* bsc = (const float*)d_in[11];
    const float* Wsh = (const float*)d_in[12]; const float* bsh = (const float*)d_in[13];
    const float* in_g= (const float*)d_in[14]; const float* in_b= (const float*)d_in[15];
    const float* ln1g= (const float*)d_in[16]; const float* ln1b= (const float*)d_in[17];
    const float* ln2g= (const float*)d_in[18]; const float* ln2b= (const float*)d_in[19];
    const float* Wq  = (const float*)d_in[20]; const float* bq  = (const float*)d_in[21];
    const float* Wk  = (const float*)d_in[22]; const float* bk  = (const float*)d_in[23];
    const float* Wv  = (const float*)d_in[24]; const float* bv  = (const float*)d_in[25];
    const float* Wo  = (const float*)d_in[26]; const float* bo  = (const float*)d_in[27];
    const float* W1  = (const float*)d_in[28]; const float* b1  = (const float*)d_in[29];
    const float* W2  = (const float*)d_in[30]; const float* b2  = (const float*)d_in[31];
    const float* outg= (const float*)d_in[32]; const float* outb= (const float*)d_in[33];
    const float* Wh1 = (const float*)d_in[34]; const float* bh1 = (const float*)d_in[35];
    const float* Wh2 = (const float*)d_in[36]; const float* bh2 = (const float*)d_in[37];
    (void)in_sizes; (void)n_in; (void)out_size;

    // ---- bump allocator over the workspace (256B aligned slices) ----
    size_t off = 0;
    auto alloc = [&](size_t bytes) -> void* {
        void* p = (char*)d_ws + off;
        off += (bytes + 255) & ~(size_t)255;
        return p;
    };
    float*     tok    = (float*)    alloc((size_t)MP_ * D_  * 4);
    float*     qb     = (float*)    alloc((size_t)MP_ * D_  * 4);
    float*     kb     = (float*)    alloc((size_t)MP_ * D_  * 4);
    float*     vb     = (float*)    alloc((size_t)MP_ * D_  * 4);
    _Float16*  x16    = (_Float16*) alloc((size_t)MP_ * D_  * 2);
    _Float16*  ff16   = (_Float16*) alloc((size_t)MP_ * FF_ * 2);
    _Float16*  feat16 = (_Float16*) alloc((size_t)MN_ * 2 * D_ * 2);
    _Float16*  hid16  = (_Float16*) alloc((size_t)MN_ * D_  * 2);
    _Float16*  wq_t   = (_Float16*) alloc((size_t)L_ * D_ * D_  * 2);
    _Float16*  wk_t   = (_Float16*) alloc((size_t)L_ * D_ * D_  * 2);
    _Float16*  wv_t   = (_Float16*) alloc((size_t)L_ * D_ * D_  * 2);
    _Float16*  wo_t   = (_Float16*) alloc((size_t)L_ * D_ * D_  * 2);
    _Float16*  w1_t   = (_Float16*) alloc((size_t)L_ * D_ * FF_ * 2);
    _Float16*  w2_t   = (_Float16*) alloc((size_t)L_ * FF_ * D_ * 2);
    _Float16*  wh1_t  = (_Float16*) alloc((size_t)2 * D_ * D_   * 2);
    if (off > ws_size) return;  // workspace too small: bail deterministically

    const int TB = 256;

    // ---- weight convert/transpose (f32 [K,N] -> f16 [N,K]) ----
    {
        dim3 gDD((D_ * D_ + TB - 1) / TB, 1, L_);
        hipLaunchKernelGGL(wcvt_kernel, gDD, dim3(TB), 0, stream, Wq, wq_t, D_, D_);
        hipLaunchKernelGGL(wcvt_kernel, gDD, dim3(TB), 0, stream, Wk, wk_t, D_, D_);
        hipLaunchKernelGGL(wcvt_kernel, gDD, dim3(TB), 0, stream, Wv, wv_t, D_, D_);
        hipLaunchKernelGGL(wcvt_kernel, gDD, dim3(TB), 0, stream, Wo, wo_t, D_, D_);
        dim3 gDF((D_ * FF_ + TB - 1) / TB, 1, L_);
        hipLaunchKernelGGL(wcvt_kernel, gDF, dim3(TB), 0, stream, W1, w1_t, D_, FF_);
        hipLaunchKernelGGL(wcvt_kernel, gDF, dim3(TB), 0, stream, W2, w2_t, FF_, D_);
        dim3 gH((2 * D_ * D_ + TB - 1) / TB, 1, 1);
        hipLaunchKernelGGL(wcvt_kernel, gH, dim3(TB), 0, stream, Wh1, wh1_t, 2 * D_, D_);
    }

    // ---- embedding + input LN (in-place f32) ----
    hipLaunchKernelGGL(embed_kernel, dim3((M_ * D_ + TB - 1) / TB), dim3(TB), 0, stream,
                       peak, spec, Wp, bp, Ws, bs, ptt, stt, remb, Wsc, bsc, Wsh, bsh, tok);
    hipLaunchKernelGGL((ln_kernel<float>), dim3(M_), dim3(32), 0, stream,
                       tok, tok, in_g, in_b, M_);

    const int MT = MP_ / 16;          // 257 row tiles
    dim3 gQKV(MT, D_ / 16);           // 257 x 10  (NB=1, A reused x3)
    dim3 gD2 (MT, D_ / 32);           // 257 x 5   (NB=2)
    dim3 gFF2(MT, FF_ / 32);          // 257 x 20  (NB=2)
    dim3 blk32(32);

    for (int l = 0; l < L_; ++l) {
        // LN1 -> f16 A operand
        hipLaunchKernelGGL((ln_kernel<_Float16>), dim3(M_), dim3(32), 0, stream,
                           tok, x16, ln1g + l * D_, ln1b + l * D_, M_);
        // fused Q/K/V projections (A tile loaded once, 3 WMMAs per k-step)
        hipLaunchKernelGGL(qkv_kernel, gQKV, blk32, 0, stream,
                           x16,
                           wq_t + (size_t)l * D_ * D_, wk_t + (size_t)l * D_ * D_,
                           wv_t + (size_t)l * D_ * D_,
                           bq + l * D_, bk + l * D_, bv + l * D_, M_,
                           qb, kb, vb);
        // attention (writes f16 into x16, reused as next GEMM's A)
        hipLaunchKernelGGL(attn_kernel, dim3((B_ * H_ * S_ + 127) / 128), dim3(128), 0, stream,
                           qb, kb, vb, peak, pad, x16);
        // O projection + residual
        hipLaunchKernelGGL((gemm_kernel<1, 2>), gD2, blk32, 0, stream,
                           x16, D_, wo_t + (size_t)l * D_ * D_, bo + l * D_, M_, D_,
                           tok, (_Float16*)nullptr, D_);
        // LN2 -> f16
        hipLaunchKernelGGL((ln_kernel<_Float16>), dim3(M_), dim3(32), 0, stream,
                           tok, x16, ln2g + l * D_, ln2b + l * D_, M_);
        // FFN1 + exact GELU -> f16
        hipLaunchKernelGGL((gemm_kernel<2, 2>), gFF2, blk32, 0, stream,
                           x16, D_, w1_t + (size_t)l * D_ * FF_, b1 + l * FF_, M_, D_,
                           (float*)nullptr, ff16, FF_);
        // FFN2 + residual
        hipLaunchKernelGGL((gemm_kernel<1, 2>), gD2, blk32, 0, stream,
                           ff16, FF_, w2_t + (size_t)l * FF_ * D_, b2 + l * D_, M_, FF_,
                           tok, (_Float16*)nullptr, D_);
    }

    // ---- output LN, feature concat, head ----
    hipLaunchKernelGGL((ln_kernel<_Float16>), dim3(M_), dim3(32), 0, stream,
                       tok, x16, outg, outb, M_);
    hipLaunchKernelGGL(feat_kernel, dim3((MN_ * 2 * D_ + TB - 1) / TB), dim3(TB), 0, stream,
                       x16, feat16);
    hipLaunchKernelGGL((gemm_kernel<2, 2>), dim3(MN_ / 16, D_ / 32), blk32, 0, stream,
                       feat16, 2 * D_, wh1_t, bh1, MN_, 2 * D_, (float*)nullptr, hid16, D_);
    hipLaunchKernelGGL(head_kernel, dim3((MN_ + TB - 1) / TB), dim3(TB), 0, stream,
                       hid16, Wh2, bh2, pad, (float*)d_out);
}
